// BlockedMLP_35038343201491
// MI455X (gfx1250) — matrix operand order repaired and checked by reading, not compiled, not run on hardware
//
#include <hip/hip_runtime.h>
#include <hip/hip_bf16.h>

typedef __attribute__((ext_vector_type(16))) _Float16 v16h;
typedef __attribute__((ext_vector_type(8)))  _Float16 v8h;
typedef __attribute__((ext_vector_type(4)))  _Float16 v4h;
typedef __attribute__((ext_vector_type(2)))  _Float16 v2h;
typedef __attribute__((ext_vector_type(8)))  float    v8f;
typedef __attribute__((ext_vector_type(4)))  int      vi4;

#define TILE_M 128
#define TILE_N 128
#define TILE_K 32
#define LDS_STRIDE 40   // 32 + 8 halves pad; keeps v8h accesses 16B-aligned, spreads banks

#if defined(__AMDGCN__) && __has_builtin(__builtin_amdgcn_global_load_async_to_lds_b128) && __has_builtin(__builtin_amdgcn_s_wait_asynccnt)
#define HAVE_ASYNC_LDS 1
#else
#define HAVE_ASYNC_LDS 0
#endif

#if HAVE_ASYNC_LDS
typedef __attribute__((address_space(1))) vi4* gp_vi4;   // global int4*
typedef __attribute__((address_space(3))) vi4* lp_vi4;   // LDS int4*
#endif

// 16B global(f16) -> LDS copy: async (ASYNCcnt-tracked, no VGPR round trip) when available.
__device__ __forceinline__ void g2l_b128(_Float16* ldst, const _Float16* gsrc) {
#if HAVE_ASYNC_LDS
  __builtin_amdgcn_global_load_async_to_lds_b128((gp_vi4)gsrc, (lp_vi4)ldst, 0, 0);
#else
  *(v8h*)ldst = *(const v8h*)gsrc;
#endif
}

// Wait for outstanding async-LDS staging, then workgroup barrier.
__device__ __forceinline__ void stage_fence() {
#if HAVE_ASYNC_LDS
  __builtin_amdgcn_s_wait_asynccnt(0);
#endif
  __syncthreads();
}

// Combine two 16B LDS chunks into one 16x32 f16 WMMA fragment register set.
__device__ __forceinline__ v16h frag16(const _Float16* p0, const _Float16* p1) {
  v8h a = *(const v8h*)p0;
  v8h b = *(const v8h*)p1;
  return __builtin_shufflevector(a, b, 0,1,2,3,4,5,6,7,8,9,10,11,12,13,14,15);
}

// Packed f32x2 -> f16x2 (v_cvt_pk_rtz_f16_f32); bit-cast __fp16 vector to _Float16 vector.
__device__ __forceinline__ v2h pk(float x, float y) {
  return __builtin_bit_cast(v2h, __builtin_amdgcn_cvt_pkrtz(x, y));
}

// 8x f32 -> 8x f16 using packed converts: 4 VALU ops not 8.
__device__ __forceinline__ v8h cvt8(const float4& a, const float4& b) {
  v2h p0 = pk(a.x, a.y), p1 = pk(a.z, a.w);
  v2h p2 = pk(b.x, b.y), p3 = pk(b.z, b.w);
  v4h lo = __builtin_shufflevector(p0, p1, 0,1,2,3);
  v4h hi = __builtin_shufflevector(p2, p3, 0,1,2,3);
  return __builtin_shufflevector(lo, hi, 0,1,2,3,4,5,6,7);
}

__device__ __forceinline__ v4h cvt4(const float4& a) {
  v2h p0 = pk(a.x, a.y), p1 = pk(a.z, a.w);
  return __builtin_shufflevector(p0, p1, 0,1,2,3);
}

// C[m,n] = sum_k A[m,k] * B[n,k]  (+bias[n], optional ReLU)
// A: M x K (f32 or f16), B: N x K weight rows (f32). WG tile 128x128, 8 waves,
// double-buffered LDS (one barrier per K-step, stage k+1 while computing k).
template<bool A_F32, bool RELU, bool OUT_F16>
__global__ __launch_bounds__(256) void gemm_wmma(
    const void* __restrict__ Ap, const float* __restrict__ Bp,
    const float* __restrict__ bias, void* __restrict__ Cp,
    int Kdim, int lda, int ldb, int ldc)
{
  __shared__ __align__(16) _Float16 lA[2][TILE_M * LDS_STRIDE];
  __shared__ __align__(16) _Float16 lB[2][TILE_N * LDS_STRIDE];
  const int t    = threadIdx.x;
  const int wave = t >> 5, lane = t & 31;
  const int wm   = wave & 1;        // 0..1 : which 64-row half
  const int wn   = wave >> 1;       // 0..3 : which 32-col quarter
  const int l15  = lane & 15, lh = lane >> 4;
  const int srow = t >> 1;          // staging: 0..127
  const int skof = (t & 1) * 16;    // staging: k offset 0 / 16
  const int gx = blockIdx.x, gy = blockIdx.y;

  v8f acc[4][2] = {};

  auto stage = [&](int buf, int k0) {
    // ---- A tile (128 x 32) ----
    if constexpr (A_F32) {
      const float* src = (const float*)Ap + (size_t)(gy*TILE_M + srow)*lda + k0 + skof;
      const float4* s4 = (const float4*)src;
      float4 a0 = s4[0], a1 = s4[1], a2 = s4[2], a3 = s4[3];
      *(v8h*)&lA[buf][srow*LDS_STRIDE + skof    ] = cvt8(a0, a1);
      *(v8h*)&lA[buf][srow*LDS_STRIDE + skof + 8] = cvt8(a2, a3);
      if (k0 + TILE_K < Kdim) __builtin_prefetch(src + TILE_K, 0, 0);
    } else {
      const _Float16* src = (const _Float16*)Ap + (size_t)(gy*TILE_M + srow)*lda + k0 + skof;
      g2l_b128(&lA[buf][srow*LDS_STRIDE + skof    ], src);
      g2l_b128(&lA[buf][srow*LDS_STRIDE + skof + 8], src + 8);
    }
    // ---- B tile: 128 weight rows x 32 k (weight rows = B columns) ----
    const float* src = Bp + (size_t)(gx*TILE_N + srow)*ldb + k0 + skof;
    const float4* s4 = (const float4*)src;
    float4 b0 = s4[0], b1 = s4[1], b2 = s4[2], b3 = s4[3];
    *(v8h*)&lB[buf][srow*LDS_STRIDE + skof    ] = cvt8(b0, b1);
    *(v8h*)&lB[buf][srow*LDS_STRIDE + skof + 8] = cvt8(b2, b3);
    if (k0 + TILE_K < Kdim) __builtin_prefetch(src + TILE_K, 0, 0);
  };

  stage(0, 0);
  int cur = 0;
  for (int k0 = 0; k0 < Kdim; k0 += TILE_K) {
    stage_fence();                       // cur buffer complete (all waves)
    if (k0 + TILE_K < Kdim) stage(cur ^ 1, k0 + TILE_K);
    // ---- fragments + WMMA from lA[cur]/lB[cur] ----
    v16h af[4], bf[2];
    #pragma unroll
    for (int i = 0; i < 4; ++i) {
      int r = (wm*64 + i*16 + l15) * LDS_STRIDE + lh*8;   // A: lane row M=l15, K chunks lh*8 / +16
      af[i] = frag16(&lA[cur][r], &lA[cur][r + 16]);
    }
    #pragma unroll
    for (int j = 0; j < 2; ++j) {
      int r = (wn*32 + j*16 + l15) * LDS_STRIDE + lh*8;   // B: lane col N=l15, K chunks lh*8 / +16
      bf[j] = frag16(&lB[cur][r], &lB[cur][r + 16]);
    }
    #pragma unroll
    for (int i = 0; i < 4; ++i)
      #pragma unroll
      for (int j = 0; j < 2; ++j)
        acc[i][j] = __builtin_amdgcn_wmma_f32_16x16x32_f16(
            false, af[i], false, bf[j], (short)0, acc[i][j], false, false);
    cur ^= 1;
  }

  // ---- epilogue: bias (+ReLU), store f16 or f32 ----
  #pragma unroll
  for (int j = 0; j < 2; ++j) {
    int col = gx*TILE_N + wn*32 + j*16 + l15;
    float bv = bias[col];
    #pragma unroll
    for (int i = 0; i < 4; ++i) {
      #pragma unroll
      for (int v = 0; v < 8; ++v) {
        int row = gy*TILE_M + wm*64 + i*16 + lh*8 + v;   // D: VGPR v -> M=v (+8 for hi lanes)
        float val = acc[i][j][v] + bv;
        if constexpr (RELU) val = val > 0.f ? val : 0.f;
        if constexpr (OUT_F16)
          ((_Float16*)Cp)[(size_t)row*ldc + col] = (_Float16)val;
        else
          ((float*)Cp)[(size_t)row*ldc + col] = val;
      }
    }
  }
}

// Block-sparse layer: y[b, rb*32+i] = relu( sum_{n in row rb} sum_j V[n,i,j]*h[b, col_n*32+j] + b2 )
// WG = 128 batch rows x one 32-wide block-row; each wave owns 16 rows, 2 N-subtiles.
// Double-buffered; gathered A block-columns staged via async global->LDS.
__global__ __launch_bounds__(256) void bsr_wmma(
    const _Float16* __restrict__ Hin, const int* __restrict__ crow,
    const int* __restrict__ cols, const float* __restrict__ vals,
    const float* __restrict__ bias, _Float16* __restrict__ Hout, int ldh)
{
  __shared__ __align__(16) _Float16 lA[2][128 * LDS_STRIDE];
  __shared__ __align__(16) _Float16 lB[2][32 * LDS_STRIDE];
  const int rb = blockIdx.x, mt = blockIdx.y;
  const int t    = threadIdx.x;
  const int wave = t >> 5, lane = t & 31;
  const int l15  = lane & 15, lh = lane >> 4;
  const int srow = t >> 1, skof = (t & 1) * 16;
  const int vr = t >> 3, vc = (t & 7) * 4;   // values staging: 4 floats/thread

  auto stage = [&](int buf, int n, int cb) {
    // A: gathered 128x32 f16 block-column of h (pure byte copy -> async to LDS)
    const _Float16* asrc = Hin + (size_t)(mt*128 + srow)*ldh + cb*32 + skof;
    g2l_b128(&lA[buf][srow*LDS_STRIDE + skof    ], asrc);
    g2l_b128(&lA[buf][srow*LDS_STRIDE + skof + 8], asrc + 8);
    // B: values[n] 32x32 f32 -> f16; B column i == values row i (contiguous in j=k)
    const float4 vv = *(const float4*)(vals + (size_t)n*1024 + vr*32 + vc);
    *(v4h*)&lB[buf][vr*LDS_STRIDE + vc] = cvt4(vv);
    __builtin_prefetch(vals + (size_t)(n+1)*1024 + vr*32 + vc, 0, 0);
  };

  v8f acc[2] = {};
  const int s = crow[rb], e = crow[rb + 1];
  if (s < e) stage(0, s, cols[s]);
  int cur = 0;
  for (int n = s; n < e; ++n) {
    stage_fence();
    if (n + 1 < e) stage(cur ^ 1, n + 1, cols[n + 1]);

    int rA = (wave*16 + l15) * LDS_STRIDE + lh*8;
    v16h af = frag16(&lA[cur][rA], &lA[cur][rA + 16]);
    #pragma unroll
    for (int j = 0; j < 2; ++j) {
      int rB = (j*16 + l15) * LDS_STRIDE + lh*8;
      v16h bfj = frag16(&lB[cur][rB], &lB[cur][rB + 16]);
      acc[j] = __builtin_amdgcn_wmma_f32_16x16x32_f16(
          false, af, false, bfj, (short)0, acc[j], false, false);
    }
    cur ^= 1;
  }
  #pragma unroll
  for (int j = 0; j < 2; ++j) {
    int col = rb*32 + j*16 + l15;
    float bv = bias[col];
    #pragma unroll
    for (int v = 0; v < 8; ++v) {
      int row = mt*128 + wave*16 + lh*8 + v;
      float val = acc[j][v] + bv;
      val = val > 0.f ? val : 0.f;
      Hout[(size_t)row*ldh + col] = (_Float16)val;
    }
  }
}

extern "C" void kernel_launch(void* const* d_in, const int* in_sizes, int n_in,
                              void* d_out, int out_size, void* d_ws, size_t ws_size,
                              hipStream_t stream) {
  const float* x    = (const float*)d_in[0];
  const float* W1   = (const float*)d_in[1];
  const float* b1   = (const float*)d_in[2];
  const int*   crow = (const int*)d_in[3];
  const int*   cols = (const int*)d_in[4];
  const float* vals = (const float*)d_in[5];
  const float* b2   = (const float*)d_in[6];
  const float* W3   = (const float*)d_in[7];
  const float* b3   = (const float*)d_in[8];
  float* out = (float*)d_out;

  const int Bv = 2048, IN = 1024, H = 2048, OUTD = 1024, RB = 64;

  // workspace: h1 (B x H f16) then h2 (B x H f16) = 16 MB total
  _Float16* h1 = (_Float16*)d_ws;
  _Float16* h2 = h1 + (size_t)Bv * H;

  dim3 blk(256);
  // layer 1: h1 = relu(x @ W1^T + b1)   M=2048 N=2048 K=1024
  gemm_wmma<true, true, true><<<dim3(H / TILE_N, Bv / TILE_M), blk, 0, stream>>>(
      x, W1, b1, h1, IN, IN, IN, H);
  // layer 2: h2 = relu(bsr(h1) + b2)
  bsr_wmma<<<dim3(RB, Bv / 128), blk, 0, stream>>>(h1, crow, cols, vals, b2, h2, H);
  // layer 3: out = h2 @ W3^T + b3       M=2048 N=1024 K=2048
  gemm_wmma<false, false, false><<<dim3(OUTD / TILE_N, Bv / TILE_M), blk, 0, stream>>>(
      h2, W3, b3, out, H, H, H, OUTD);
}
